// RT_Clearing_17652315586613
// MI455X (gfx1250) — matrix operand order repaired and checked by reading, not compiled, not run on hardware
//
#include <hip/hip_runtime.h>
#include <hip/hip_bf16.h>

// ---------------------------------------------------------------------------
// Problem constants (from the reference)
// ---------------------------------------------------------------------------
#define N_G     100
#define N_LOAD  200
#define N_LINE  500
#define N_BUS   300
#define N_W     50
#define N_SCEN  128
#define C_VIOL  20000.0f
#define N_ITERS 600
#define POWER_ITERS 50

// n = 3G + D = 500  -> pad 512 ; m = 2G + 2L + 1 = 1201 -> pad 1216
#define N_DIM   500
#define M_DIM   1201
#define N_PAD   512
#define M_PAD   1216
#define KT_X    38      // M_PAD/32   k-chunks for  y@K
#define NT_X    32      // N_PAD/16   n-tiles  for  y@K
#define KT_Y    16      // N_PAD/32   k-chunks for  xb@K^T
#define NT_Y    76      // M_PAD/16   n-tiles  for  xb@K^T
#define ST_TILES 8      // N_SCEN/16

typedef __attribute__((ext_vector_type(16))) _Float16 v16h;
typedef __attribute__((ext_vector_type(8)))  _Float16 v8h;
typedef __attribute__((ext_vector_type(8)))  float    v8f;

// ---------------------------------------------------------------------------
// Setup: Hg/Hl/Hw = PTDF @ node_* (dense small matmuls, one thread / element)
// ---------------------------------------------------------------------------
__global__ void k_build_H(const float* __restrict__ PTDF,
                          const float* __restrict__ nodeG,
                          const float* __restrict__ nodeL,
                          const float* __restrict__ nodeW,
                          float* __restrict__ Hg, float* __restrict__ Hl,
                          float* __restrict__ Hw)
{
    int id = blockIdx.x * blockDim.x + threadIdx.x;
    const int total = N_LINE * (N_G + N_LOAD + N_W);
    if (id >= total) return;
    int l = id / (N_G + N_LOAD + N_W);
    int j = id % (N_G + N_LOAD + N_W);
    const float* prow = PTDF + l * N_BUS;
    float acc = 0.f;
    if (j < N_G) {
        for (int b = 0; b < N_BUS; ++b) acc += prow[b] * nodeG[b * N_G + j];
        Hg[l * N_G + j] = acc;
    } else if (j < N_G + N_LOAD) {
        int jj = j - N_G;
        for (int b = 0; b < N_BUS; ++b) acc += prow[b] * nodeL[b * N_LOAD + jj];
        Hl[l * N_LOAD + jj] = acc;
    } else {
        int jj = j - N_G - N_LOAD;
        for (int b = 0; b < N_BUS; ++b) acc += prow[b] * nodeW[b * N_W + jj];
        Hw[l * N_W + jj] = acc;
    }
}

// ---------------------------------------------------------------------------
// Build padded K (M_PAD x N_PAD, f32) and c (N_PAD)
// ---------------------------------------------------------------------------
__global__ void k_build_K(const float* __restrict__ Hg,
                          const float* __restrict__ Hl,
                          float* __restrict__ K, float* __restrict__ cvec)
{
    int id = blockIdx.x * blockDim.x + threadIdx.x;
    if (id >= M_PAD * N_PAD) return;
    int r = id / N_PAD;
    int j = id % N_PAD;
    float val = 0.f;
    if (j < N_DIM && r < M_DIM) {
        if (r < 2 * N_G) {                     // [I 0 0 0] ; [0 I 0 0]
            val = (j == r) ? 1.f : 0.f;
        } else if (r < 1200) {                 // +/- [Hg -Hg -Hg Hl]
            int  l   = (r < 700) ? r - 200 : r - 700;
            float sg = (r < 700) ? 1.f : -1.f;
            float h;
            if      (j < 100) h =  Hg[l * N_G + j];
            else if (j < 200) h = -Hg[l * N_G + (j - 100)];
            else if (j < 300) h = -Hg[l * N_G + (j - 200)];
            else              h =  Hl[l * N_LOAD + (j - 300)];
            val = sg * h;
        } else {                               // r == 1200 : [1 -1 -1 1]
            val = (j < 100) ? 1.f : (j < 300 ? -1.f : 1.f);
        }
    }
    K[id] = val;
    if (id < N_PAD)
        cvec[id] = (id >= 2 * N_G && id < N_DIM) ? C_VIOL : 0.f;
}

// ---------------------------------------------------------------------------
// Build q (N_SCEN x M_PAD, f32). Reserves computed inline (cheap).
// ---------------------------------------------------------------------------
__global__ void k_build_q(const float* __restrict__ p_da,
                          const float* __restrict__ r_up_hat,
                          const float* __restrict__ r_down_hat,
                          const float* __restrict__ f_up_hat,
                          const float* __restrict__ f_down_hat,
                          const float* __restrict__ w,
                          const float* __restrict__ Pmax,
                          const float* __restrict__ Hw,
                          float* __restrict__ q)
{
    int id = blockIdx.x * blockDim.x + threadIdx.x;
    if (id >= N_SCEN * M_PAD) return;
    int s   = id / M_PAD;
    int col = id % M_PAD;
    float val = 0.f;
    if (col < N_G) {
        val = fmaxf(fminf(r_up_hat[col], Pmax[col] - p_da[col]), 0.f);
    } else if (col < 2 * N_G) {
        int g = col - N_G;
        val = fmaxf(fminf(r_down_hat[g], p_da[g]), 0.f);
    } else if (col < 1200) {
        int l = (col < 700) ? col - 200 : col - 700;
        float wh = 0.f;
        const float* ws_ = w + s * N_W;
        const float* hw  = Hw + l * N_W;
        for (int k = 0; k < N_W; ++k) wh += ws_[k] * hw[k];
        val = (col < 700) ? fmaxf(f_up_hat[l], 0.f) - wh
                          : fmaxf(f_down_hat[l], 0.f) + wh;
    } else if (col == 1200) {
        float sw = 0.f;
        const float* ws_ = w + s * N_W;
        for (int k = 0; k < N_W; ++k) sw += ws_[k];
        val = -sw;
    }
    q[id] = val;
}

// ---------------------------------------------------------------------------
// Pack K into f16 WMMA B-operand lane layout.
//  B tile is 32(K) x 16(N); lane L holds column N=L%16, K-half L/16;
//  half h of the lane's v16h = element (8*(L/16)+(h&7)+16*(h>>3) , L%16) of the tile.
//  mode 0: B = K    (k over M_PAD, n over N_PAD)   -> pkB [KT_X][NT_X][32][16]
//  mode 1: B = K^T  (k over N_PAD, n over M_PAD)   -> pkBT[KT_Y][NT_Y][32][16]
// ---------------------------------------------------------------------------
__global__ void k_pack(const float* __restrict__ K, _Float16* __restrict__ dst,
                       int mode)
{
    int id = blockIdx.x * blockDim.x + threadIdx.x;
    const int total = (mode == 0 ? KT_X * NT_X : KT_Y * NT_Y) * 32 * 16;
    if (id >= total) return;
    int h    = id & 15;
    int lane = (id >> 4) & 31;
    int tile = id >> 9;
    int ntc  = (mode == 0) ? NT_X : NT_Y;
    int nt   = tile % ntc;
    int kt   = tile / ntc;
    int kk   = kt * 32 + 8 * (lane >> 4) + (h & 7) + 16 * (h >> 3);
    int nn   = nt * 16 + (lane & 15);
    float v  = (mode == 0) ? K[kk * N_PAD + nn]   // B = K
                           : K[nn * N_PAD + kk];  // B = K^T
    dst[id] = (_Float16)v;
}

// ---------------------------------------------------------------------------
// Power iteration (single workgroup): v <- K^T(Kv)/||.|| ; step = 0.9/sqrt(s)
// ---------------------------------------------------------------------------
__global__ __launch_bounds__(1024) void k_power(const float* __restrict__ K,
                                                float* __restrict__ stepOut)
{
    __shared__ float v[N_PAD];
    __shared__ float u[N_PAD];
    __shared__ float t[M_PAD];
    __shared__ float red[1024];
    int tid = threadIdx.x;

    for (int j = tid; j < N_PAD; j += 1024)
        v[j] = (j < N_DIM) ? rsqrtf((float)N_DIM) : 0.f;
    __syncthreads();

    float nrm = 1.f;
    for (int it = 0; it < POWER_ITERS; ++it) {
        for (int r = tid; r < M_PAD; r += 1024) {
            const float* kr = K + r * N_PAD;
            float a = 0.f;
            for (int j = 0; j < N_PAD; ++j) a += kr[j] * v[j];
            t[r] = a;
        }
        __syncthreads();
        for (int j = tid; j < N_PAD; j += 1024) {
            float a = 0.f;
            for (int r = 0; r < M_PAD; ++r) a += K[r * N_PAD + j] * t[r];
            u[j] = a;
        }
        __syncthreads();
        float p = 0.f;
        for (int j = tid; j < N_PAD; j += 1024) p += u[j] * u[j];
        red[tid] = p;
        __syncthreads();
        for (int off = 512; off > 0; off >>= 1) {
            if (tid < off) red[tid] += red[tid + off];
            __syncthreads();
        }
        nrm = sqrtf(red[0]);
        __syncthreads();
        for (int j = tid; j < N_PAD; j += 1024) v[j] = u[j] / nrm;
        __syncthreads();
    }
    if (tid == 0) stepOut[0] = 0.9f / sqrtf(nrm);
}

// ---------------------------------------------------------------------------
// Zero-initialise PDHG state
// ---------------------------------------------------------------------------
__global__ void k_init(float* __restrict__ x, float* __restrict__ y,
                       _Float16* __restrict__ xbh, _Float16* __restrict__ yh)
{
    int id = blockIdx.x * blockDim.x + threadIdx.x;
    const int NX = N_SCEN * N_PAD, NY = N_SCEN * M_PAD;
    if (id < NX) { x[id] = 0.f; xbh[id] = (_Float16)0.f; }
    else if (id < NX + NY) { int j = id - NX; y[j] = 0.f; yh[j] = (_Float16)0.f; }
}

// ---------------------------------------------------------------------------
// X-step:  Z = y@K (WMMA);  x1 = relu(x - step*(c+Z));  xb = 2x1 - x
//   one wave32 per 16x16 D-tile; block = 4 waves; grid = 64 blocks (256 tiles)
// ---------------------------------------------------------------------------
__global__ __launch_bounds__(128) void k_xstep(
        const _Float16* __restrict__ yh, const _Float16* __restrict__ pkB,
        const float* __restrict__ cvec, const float* __restrict__ stepPtr,
        float* __restrict__ x, _Float16* __restrict__ xbh)
{
    int wave = threadIdx.x >> 5;
    int lane = threadIdx.x & 31;
    int tile = blockIdx.x * 4 + wave;          // 0..255
    int st = tile >> 5;                        // scenario tile 0..7
    int nt = tile & 31;                        // n tile 0..31
    int half = lane >> 4;
    int rowA = st * 16 + (lane & 15);
    const _Float16* yrow = yh + (size_t)rowA * M_PAD + half * 8;

    v8f acc = {};
    for (int kc = 0; kc < KT_X; ++kc) {
        v8h lo = *(const v8h*)(yrow + kc * 32);
        v8h hi = *(const v8h*)(yrow + kc * 32 + 16);
        v16h a = __builtin_shufflevector(lo, hi, 0,1,2,3,4,5,6,7,
                                                 8,9,10,11,12,13,14,15);
        v16h b = *(const v16h*)(pkB + ((((size_t)kc * NT_X + nt) * 32) + lane) * 16);
        acc = __builtin_amdgcn_wmma_f32_16x16x32_f16(
                  false, a, false, b, (short)0, acc, false, false);
    }

    float step = stepPtr[0];
    int col  = nt * 16 + (lane & 15);
    int row0 = st * 16 + half * 8;
    float cc = cvec[col];
    #pragma unroll
    for (int v = 0; v < 8; ++v) {
        int idx = (row0 + v) * N_PAD + col;
        float xo = x[idx];
        float x1 = fmaxf(xo - step * (cc + acc[v]), 0.f);
        x[idx]   = x1;
        xbh[idx] = (_Float16)(2.f * x1 - xo);
    }
}

// ---------------------------------------------------------------------------
// Y-step:  Z2 = xb@K^T (WMMA);  y1 = y + step*(Z2 - q); project ineq rows
//   608 tiles; block = 4 waves; grid = 152 blocks
// ---------------------------------------------------------------------------
__global__ __launch_bounds__(128) void k_ystep(
        const _Float16* __restrict__ xbh, const _Float16* __restrict__ pkBT,
        const float* __restrict__ q, const float* __restrict__ stepPtr,
        float* __restrict__ y, _Float16* __restrict__ yh)
{
    int wave = threadIdx.x >> 5;
    int lane = threadIdx.x & 31;
    int tile = blockIdx.x * 4 + wave;          // 0..607
    int st = tile / NT_Y;                      // scenario tile 0..7
    int nt = tile % NT_Y;                      // m tile 0..75
    int half = lane >> 4;
    int rowA = st * 16 + (lane & 15);
    const _Float16* xrow = xbh + (size_t)rowA * N_PAD + half * 8;

    v8f acc = {};
    for (int kc = 0; kc < KT_Y; ++kc) {
        v8h lo = *(const v8h*)(xrow + kc * 32);
        v8h hi = *(const v8h*)(xrow + kc * 32 + 16);
        v16h a = __builtin_shufflevector(lo, hi, 0,1,2,3,4,5,6,7,
                                                 8,9,10,11,12,13,14,15);
        v16h b = *(const v16h*)(pkBT + ((((size_t)kc * NT_Y + nt) * 32) + lane) * 16);
        acc = __builtin_amdgcn_wmma_f32_16x16x32_f16(
                  false, a, false, b, (short)0, acc, false, false);
    }

    float step = stepPtr[0];
    int col  = nt * 16 + (lane & 15);
    int row0 = st * 16 + half * 8;
    #pragma unroll
    for (int v = 0; v < 8; ++v) {
        int idx = (row0 + v) * M_PAD + col;
        float yo = y[idx];
        float y1 = yo + step * (acc[v] - q[idx]);
        if (col < M_DIM - 1) y1 = fmaxf(y1, 0.f);   // ineq rows only
        y[idx] = y1;
        yh[idx] = (_Float16)y1;
    }
}

// ---------------------------------------------------------------------------
// Final cost: out[s] = C_VIOL * sum(x[s, 200:500])
// ---------------------------------------------------------------------------
__global__ void k_cost(const float* __restrict__ x, float* __restrict__ out)
{
    __shared__ float red[64];
    int s = blockIdx.x;
    float p = 0.f;
    for (int j = 2 * N_G + threadIdx.x; j < N_DIM; j += 64)
        p += x[s * N_PAD + j];
    red[threadIdx.x] = p;
    __syncthreads();
    for (int off = 32; off > 0; off >>= 1) {
        if (threadIdx.x < off) red[threadIdx.x] += red[threadIdx.x + off];
        __syncthreads();
    }
    if (threadIdx.x == 0) out[s] = C_VIOL * red[0];
}

// ---------------------------------------------------------------------------
// Host driver
// ---------------------------------------------------------------------------
extern "C" void kernel_launch(void* const* d_in, const int* in_sizes, int n_in,
                              void* d_out, int out_size, void* d_ws, size_t ws_size,
                              hipStream_t stream)
{
    const float* p_da       = (const float*)d_in[0];
    const float* r_up_hat   = (const float*)d_in[1];
    const float* r_down_hat = (const float*)d_in[2];
    const float* f_up_hat   = (const float*)d_in[3];
    const float* f_down_hat = (const float*)d_in[4];
    const float* w_err      = (const float*)d_in[5];
    const float* Pmax       = (const float*)d_in[6];
    const float* PTDF       = (const float*)d_in[7];
    const float* nodeG      = (const float*)d_in[8];
    const float* nodeL      = (const float*)d_in[9];
    const float* nodeW      = (const float*)d_in[10];
    float* out = (float*)d_out;

    // workspace carve-out (256B aligned)
    char* base = (char*)d_ws;
    size_t off = 0;
    auto alloc = [&](size_t bytes) {
        size_t o = off;
        off = (off + bytes + 255) & ~(size_t)255;
        return base + o;
    };
    float*    Hg   = (float*)alloc(sizeof(float) * N_LINE * N_G);
    float*    Hl   = (float*)alloc(sizeof(float) * N_LINE * N_LOAD);
    float*    Hw   = (float*)alloc(sizeof(float) * N_LINE * N_W);
    float*    K    = (float*)alloc(sizeof(float) * M_PAD * N_PAD);
    float*    cvec = (float*)alloc(sizeof(float) * N_PAD);
    float*    q    = (float*)alloc(sizeof(float) * N_SCEN * M_PAD);
    float*    x    = (float*)alloc(sizeof(float) * N_SCEN * N_PAD);
    float*    y    = (float*)alloc(sizeof(float) * N_SCEN * M_PAD);
    float*    step = (float*)alloc(sizeof(float) * 8);
    _Float16* pkB  = (_Float16*)alloc(sizeof(_Float16) * KT_X * NT_X * 32 * 16);
    _Float16* pkBT = (_Float16*)alloc(sizeof(_Float16) * KT_Y * NT_Y * 32 * 16);
    _Float16* xbh  = (_Float16*)alloc(sizeof(_Float16) * N_SCEN * N_PAD);
    _Float16* yh   = (_Float16*)alloc(sizeof(_Float16) * N_SCEN * M_PAD);

    // ---- setup ----
    {
        int tot = N_LINE * (N_G + N_LOAD + N_W);
        k_build_H<<<(tot + 255) / 256, 256, 0, stream>>>(PTDF, nodeG, nodeL,
                                                         nodeW, Hg, Hl, Hw);
    }
    k_build_K<<<(M_PAD * N_PAD + 255) / 256, 256, 0, stream>>>(Hg, Hl, K, cvec);
    k_build_q<<<(N_SCEN * M_PAD + 255) / 256, 256, 0, stream>>>(
        p_da, r_up_hat, r_down_hat, f_up_hat, f_down_hat, w_err, Pmax, Hw, q);
    {
        int tot = KT_X * NT_X * 32 * 16;
        k_pack<<<(tot + 255) / 256, 256, 0, stream>>>(K, pkB, 0);
        tot = KT_Y * NT_Y * 32 * 16;
        k_pack<<<(tot + 255) / 256, 256, 0, stream>>>(K, pkBT, 1);
    }
    k_power<<<1, 1024, 0, stream>>>(K, step);
    {
        int tot = N_SCEN * (N_PAD + M_PAD);
        k_init<<<(tot + 255) / 256, 256, 0, stream>>>(x, y, xbh, yh);
    }

    // ---- PDHG iterations ----
    for (int it = 0; it < N_ITERS; ++it) {
        k_xstep<<<(ST_TILES * NT_X) / 4, 128, 0, stream>>>(yh, pkB, cvec, step,
                                                           x, xbh);
        k_ystep<<<(ST_TILES * NT_Y) / 4, 128, 0, stream>>>(xbh, pkBT, q, step,
                                                           y, yh);
    }

    // ---- cost ----
    k_cost<<<N_SCEN, 64, 0, stream>>>(x, out);
}
